// MultiHeadAttention_42107859370627
// MI455X (gfx1250) — matrix operand order repaired and checked by reading, not compiled, not run on hardware
//
#include <hip/hip_runtime.h>

// MHA: B=2, S=2048, D=1024, H=16, HD=64. fp32 in/out, bf16 WMMA internally.
// QKV/O-proj: 16x64-per-wave WMMA GEMMs (4x A-fragment reuse, batched B loads
// so the compiler can issue one load clause + progressive waitcnts).
// Attention: flash-style online softmax; K/V tiles staged into LDS via the
// CDNA5 Tensor Data Mover (tensor_load_to_lds + s_wait_tensorcnt).

#define BATCH 2
#define SEQ   2048
#define DMODEL 1024
#define NHEAD 16
#define HDIM  64
#define MASKV (-1.7014118e38f)   // finfo(f32).min / 2

typedef __attribute__((ext_vector_type(16))) __bf16 v16bf;
typedef __attribute__((ext_vector_type(8)))  __bf16 v8bf;
typedef __attribute__((ext_vector_type(8)))  float  v8f;
typedef __attribute__((ext_vector_type(4)))  unsigned int u32x4;
typedef __attribute__((ext_vector_type(8)))  int          i32x8;
typedef __attribute__((ext_vector_type(4)))  int          i32x4;

#if defined(__has_builtin)
#if __has_builtin(__builtin_amdgcn_tensor_load_to_lds) && \
    __has_builtin(__builtin_amdgcn_s_wait_tensorcnt)
#define HAVE_TDM 1
#endif
#endif

static __device__ inline v16bf cat8(v8bf lo, v8bf hi) {
  return __builtin_shufflevector(lo, hi, 0,1,2,3,4,5,6,7,8,9,10,11,12,13,14,15);
}

// A fragment (16x32 bf16, M x K). p = row base (+k0 already added).
static __device__ inline v16bf load_a16x32(const __bf16* p, int half) {
  v8bf lo = *(const v8bf*)(p + 8 * half);
  v8bf hi = *(const v8bf*)(p + 16 + 8 * half);
  return cat8(lo, hi);
}

// B fragment (32x16 bf16, K x N), stored as B^T rows (contiguous in K).
static __device__ inline v16bf load_b32x16(const __bf16* p, int half) {
  const __bf16* q = p + 8 * half;
  v8bf lo = *(const v8bf*)(q);
  v8bf hi = *(const v8bf*)(q + 16);
  return cat8(lo, hi);
}

static __device__ inline v8f wmma_bf16(v16bf a, v16bf b, v8f c) {
  return __builtin_amdgcn_wmma_f32_16x16x32_bf16(false, a, false, b, (short)0, c,
                                                 false, false);
}

#ifdef HAVE_TDM
// 2D tile DMA: global (row-major, elem=2B) -> LDS (tile_w-contiguous rows).
// Descriptor packing per CDNA5 ISA sec.8.
static __device__ inline void tdm_load_2d_bf16(unsigned lds_off,
                                               const void* gptr,
                                               unsigned tile_w, unsigned tile_h,
                                               unsigned tensor_w, unsigned tensor_h,
                                               unsigned row_stride_elems) {
  const unsigned long long ga = (unsigned long long)gptr;
  u32x4 g0;
  g0[0] = 1u;                                   // count=1, user mode
  g0[1] = lds_off;                              // lds_addr
  g0[2] = (unsigned)(ga & 0xffffffffu);         // global_addr[31:0]
  g0[3] = (unsigned)((ga >> 32) & 0x01ffffffu)  // global_addr[56:32]
          | (2u << 30);                         // type=2 ("image")
  i32x8 g1;
  g1[0] = 0x00010000;                           // data_size=1 (2 bytes)
  g1[1] = (int)((tensor_w & 0xffffu) << 16);    // tensor_dim0[15:0]
  g1[2] = (int)(((tensor_w >> 16) & 0xffffu) | ((tensor_h & 0xffffu) << 16));
  g1[3] = (int)(((tensor_h >> 16) & 0xffffu) | ((tile_w & 0xffffu) << 16));
  g1[4] = (int)(tile_h & 0xffffu);              // tile_dim1 (tile_dim2=0)
  g1[5] = (int)row_stride_elems;                // tensor_dim0_stride[31:0]
  g1[6] = 0;                                    // stride[47:32], dim1_stride lo
  g1[7] = 0;
  i32x4 gz = {0, 0, 0, 0};
#if __clang_major__ >= 23
  i32x8 gz8 = {0, 0, 0, 0, 0, 0, 0, 0};
  __builtin_amdgcn_tensor_load_to_lds(g0, g1, gz, gz, gz8, 0);
#else
  __builtin_amdgcn_tensor_load_to_lds(g0, g1, gz, gz, 0);
#endif
}
#endif

// ---------------------------------------------------------------- convert ----
__global__ void cvt_f32_bf16(const float* __restrict__ src,
                             __bf16* __restrict__ dst, int n) {
  for (int i = blockIdx.x * blockDim.x + threadIdx.x; i < n;
       i += gridDim.x * blockDim.x)
    dst[i] = (__bf16)src[i];
}

// ------------------------------------------------------------------- QKV -----
// One wave per 16(token) x 64(feature) tile == one full head slice.
__global__ __launch_bounds__(32)
void qkv_kernel(const __bf16* __restrict__ x_bf,
                const __bf16* __restrict__ wq_bf,
                const __bf16* __restrict__ wk_bf,
                const __bf16* __restrict__ wv_bf,
                __bf16* __restrict__ q_bf,
                __bf16* __restrict__ k_bf,
                __bf16* __restrict__ vt_bf) {
  const int lane = threadIdx.x;
  const int half = lane >> 4;
  const int lp   = lane & 15;
  const int tm = blockIdx.x;           // token tile   (B*S/16)
  const int h  = blockIdx.y;           // head == 64-wide N tile (D/64)
  const int z  = blockIdx.z;           // 0=Q 1=K 2=V

  const __bf16* W = (z == 0) ? wq_bf : (z == 1) ? wk_bf : wv_bf;

  const __bf16* arow = x_bf + (size_t)(tm * 16 + lp) * DMODEL;
  const __bf16* brow0 = W + (size_t)(h * HDIM + lp) * DMODEL;

  v8f acc[4];
#pragma unroll
  for (int t = 0; t < 4; ++t) acc[t] = (v8f){};

#pragma unroll 2
  for (int k0 = 0; k0 < DMODEL; k0 += 32) {
    if (k0 + 32 < DMODEL) {
      __builtin_prefetch(arow + k0 + 32, 0, 0);
      __builtin_prefetch(brow0 + k0 + 32, 0, 0);
    }
    // Batch all loads so they issue as one clause with partial waits.
    const v16bf a  = load_a16x32(arow + k0, half);
    const v16bf b0 = load_b32x16(brow0 + (size_t)( 0) * DMODEL + k0, half);
    const v16bf b1 = load_b32x16(brow0 + (size_t)(16) * DMODEL + k0, half);
    const v16bf b2 = load_b32x16(brow0 + (size_t)(32) * DMODEL + k0, half);
    const v16bf b3 = load_b32x16(brow0 + (size_t)(48) * DMODEL + k0, half);
    acc[0] = wmma_bf16(a, b0, acc[0]);
    acc[1] = wmma_bf16(a, b1, acc[1]);
    acc[2] = wmma_bf16(a, b2, acc[2]);
    acc[3] = wmma_bf16(a, b3, acc[3]);
  }

  if (z == 2) {
    // V transposed (b,h,hd,s): lane's 8 rows = 8 consecutive tokens.
    const int t0 = tm * 16 + 8 * half;
    const int bb = t0 / SEQ, s0 = t0 % SEQ;
#pragma unroll
    for (int t = 0; t < 4; ++t) {
      const int hd = t * 16 + lp;
      v8bf v;
#pragma unroll
      for (int r = 0; r < 8; ++r) v[r] = (__bf16)acc[t][r];
      *(v8bf*)(vt_bf + ((size_t)(bb * NHEAD + h) * HDIM + hd) * SEQ + s0) = v;
    }
  } else {
    const float sc = (z == 0) ? 0.125f : 1.0f;   // HD^-0.5 folded into Q
    __bf16* dst = (z == 0) ? q_bf : k_bf;
#pragma unroll
    for (int t = 0; t < 4; ++t) {
      const int hd = t * 16 + lp;
#pragma unroll
      for (int r = 0; r < 8; ++r) {
        const int tok = tm * 16 + r + 8 * half;
        const int bb = tok / SEQ, s = tok % SEQ;
        dst[((size_t)(bb * NHEAD + h) * SEQ + s) * HDIM + hd] =
            (__bf16)(acc[t][r] * sc);
      }
    }
  }
}

// ------------------------------------------------------------- attention ----
__global__ __launch_bounds__(32)
void attn_kernel(const __bf16* __restrict__ q_bf,
                 const __bf16* __restrict__ k_bf,
                 const __bf16* __restrict__ vt_bf,
                 const unsigned char* __restrict__ amask,
                 __bf16* __restrict__ ctx_bf) {
  const int lane = threadIdx.x;
  const int half = lane >> 4;
  const int lp   = lane & 15;
  const int q0 = blockIdx.x * 16;
  const int bh = blockIdx.y;            // b*NHEAD + h
  const int b  = bh / NHEAD;
  const int h  = bh % NHEAD;

  __shared__ __bf16 Klds[32 * HDIM];    // 32 keys x 64 hd  (4 KB)
  __shared__ __bf16 Vlds[HDIM * 32];    // 64 hd   x 32 keys (4 KB)
  __shared__ __bf16 Plds[16][40];       // P transpose staging

  const __bf16* qrow = q_bf + ((size_t)bh * SEQ + q0 + lp) * HDIM;
  const v16bf qa0 = load_a16x32(qrow, half);        // hd 0..31
  const v16bf qa1 = load_a16x32(qrow + 32, half);   // hd 32..63

  v8f acc[4];
  float m[8], l[8];
#pragma unroll
  for (int t = 0; t < 4; ++t) acc[t] = (v8f){};
#pragma unroll
  for (int r = 0; r < 8; ++r) { m[r] = -__builtin_inff(); l[r] = 0.0f; }

  const int kend = q0 + 16;             // causal bound (exclusive)
  for (int kb = 0; kb < kend; kb += 32) {
#ifdef HAVE_TDM
    // ---- TDM: stage K (32x64) and V^T (64x32) tiles into LDS ----
    tdm_load_2d_bf16((unsigned)(size_t)&Klds[0],
                     k_bf + ((size_t)bh * SEQ + kb) * HDIM,
                     /*tile_w=*/HDIM, /*tile_h=*/32,
                     /*tensor_w=*/HDIM, /*tensor_h=*/SEQ - kb,
                     /*row_stride=*/HDIM);
    tdm_load_2d_bf16((unsigned)(size_t)&Vlds[0],
                     vt_bf + (size_t)bh * HDIM * SEQ + kb,
                     /*tile_w=*/32, /*tile_h=*/HDIM,
                     /*tensor_w=*/SEQ - kb, /*tensor_h=*/HDIM,
                     /*row_stride=*/SEQ);
    __builtin_amdgcn_s_wait_tensorcnt(0);
    const __bf16* kr0 = &Klds[(size_t)lp * HDIM];
    const __bf16* kr1 = &Klds[(size_t)(16 + lp) * HDIM];
#else
    const __bf16* kr0 = k_bf + ((size_t)bh * SEQ + kb + lp) * HDIM;
    const __bf16* kr1 = k_bf + ((size_t)bh * SEQ + kb + 16 + lp) * HDIM;
#endif
    // ---- scores: batch the 4 K fragments, then 4 WMMAs ----
    const v16bf bk00 = load_b32x16(kr0, half);
    const v16bf bk01 = load_b32x16(kr0 + 32, half);
    const v16bf bk10 = load_b32x16(kr1, half);
    const v16bf bk11 = load_b32x16(kr1 + 32, half);
    v8f s0 = {}, s1 = {};
    s0 = wmma_bf16(qa0, bk00, s0);
    s0 = wmma_bf16(qa1, bk01, s0);
    s1 = wmma_bf16(qa0, bk10, s1);
    s1 = wmma_bf16(qa1, bk11, s1);

    // ---- mask + online softmax (row = r + 8*half, col = lane in half) ----
    const int key0 = kb + lp, key1 = kb + 16 + lp;
    const bool av0 = amask[b * SEQ + key0] != 0;
    const bool av1 = amask[b * SEQ + key1] != 0;
    float p0a[8], p1a[8], scr[8];
#pragma unroll
    for (int r = 0; r < 8; ++r) {
      const int qg = q0 + r + 8 * half;
      float e0 = ((key0 > qg) || !av0) ? MASKV : s0[r];
      float e1 = ((key1 > qg) || !av1) ? MASKV : s1[r];
      float mc = fmaxf(e0, e1);
#pragma unroll
      for (int off = 1; off < 16; off <<= 1)
        mc = fmaxf(mc, __shfl_xor(mc, off, 32));
      const float mn = fmaxf(m[r], mc);
      const float p0 = __expf(e0 - mn);
      const float p1 = __expf(e1 - mn);
      float rs = p0 + p1;
#pragma unroll
      for (int off = 1; off < 16; off <<= 1) rs += __shfl_xor(rs, off, 32);
      const float sc = __expf(m[r] - mn);
      l[r] = l[r] * sc + rs;
      m[r] = mn;
      scr[r] = sc;
      p0a[r] = p0;
      p1a[r] = p1;
    }
#pragma unroll
    for (int t = 0; t < 4; ++t)
#pragma unroll
      for (int r = 0; r < 8; ++r) acc[t][r] *= scr[r];

    // ---- transpose P (C-layout -> A-layout) through LDS ----
    __syncthreads();
#pragma unroll
    for (int r = 0; r < 8; ++r) {
      Plds[r + 8 * half][lp]      = (__bf16)p0a[r];
      Plds[r + 8 * half][16 + lp] = (__bf16)p1a[r];
    }
    __syncthreads();
    const v16bf pa = load_a16x32(&Plds[lp][0], half);

    // ---- ctx += P x V : batch the 4 V fragments, then 4 WMMAs ----
#ifdef HAVE_TDM
    const v16bf bv0 = load_b32x16(&Vlds[(size_t)( 0 + lp) * 32], half);
    const v16bf bv1 = load_b32x16(&Vlds[(size_t)(16 + lp) * 32], half);
    const v16bf bv2 = load_b32x16(&Vlds[(size_t)(32 + lp) * 32], half);
    const v16bf bv3 = load_b32x16(&Vlds[(size_t)(48 + lp) * 32], half);
#else
    const __bf16* vb = vt_bf + (size_t)bh * HDIM * SEQ + kb;
    const v16bf bv0 = load_b32x16(vb + (size_t)( 0 + lp) * SEQ, half);
    const v16bf bv1 = load_b32x16(vb + (size_t)(16 + lp) * SEQ, half);
    const v16bf bv2 = load_b32x16(vb + (size_t)(32 + lp) * SEQ, half);
    const v16bf bv3 = load_b32x16(vb + (size_t)(48 + lp) * SEQ, half);
#endif
    acc[0] = wmma_bf16(pa, bv0, acc[0]);
    acc[1] = wmma_bf16(pa, bv1, acc[1]);
    acc[2] = wmma_bf16(pa, bv2, acc[2]);
    acc[3] = wmma_bf16(pa, bv3, acc[3]);
    __syncthreads();
  }

  // ---- normalize + store ctx as (b, s, h*64+hd) bf16 ----
#pragma unroll
  for (int t = 0; t < 4; ++t)
#pragma unroll
    for (int r = 0; r < 8; ++r) {
      const float inv = (l[r] > 0.0f) ? (1.0f / l[r]) : 0.0f;
      const size_t idx =
          ((size_t)b * SEQ + q0 + r + 8 * half) * DMODEL + h * HDIM + t * 16 + lp;
      ctx_bf[idx] = (__bf16)(acc[t][r] * inv);
    }
}

// ------------------------------------------------------------ out proj ------
__global__ __launch_bounds__(32)
void oproj_kernel(const __bf16* __restrict__ ctx_bf,
                  const __bf16* __restrict__ wo_bf,
                  const float* __restrict__ bo,
                  float* __restrict__ out) {
  const int lane = threadIdx.x;
  const int half = lane >> 4;
  const int lp   = lane & 15;
  const int tm = blockIdx.x;
  const int tn = blockIdx.y;            // 64-wide N tile

  const __bf16* arow = ctx_bf + (size_t)(tm * 16 + lp) * DMODEL;
  const __bf16* brow0 = wo_bf + (size_t)(tn * 64 + lp) * DMODEL;

  v8f acc[4];
#pragma unroll
  for (int t = 0; t < 4; ++t) acc[t] = (v8f){};

#pragma unroll 2
  for (int k0 = 0; k0 < DMODEL; k0 += 32) {
    if (k0 + 32 < DMODEL) {
      __builtin_prefetch(arow + k0 + 32, 0, 0);
      __builtin_prefetch(brow0 + k0 + 32, 0, 0);
    }
    const v16bf a  = load_a16x32(arow + k0, half);
    const v16bf b0 = load_b32x16(brow0 + (size_t)( 0) * DMODEL + k0, half);
    const v16bf b1 = load_b32x16(brow0 + (size_t)(16) * DMODEL + k0, half);
    const v16bf b2 = load_b32x16(brow0 + (size_t)(32) * DMODEL + k0, half);
    const v16bf b3 = load_b32x16(brow0 + (size_t)(48) * DMODEL + k0, half);
    acc[0] = wmma_bf16(a, b0, acc[0]);
    acc[1] = wmma_bf16(a, b1, acc[1]);
    acc[2] = wmma_bf16(a, b2, acc[2]);
    acc[3] = wmma_bf16(a, b3, acc[3]);
  }

#pragma unroll
  for (int t = 0; t < 4; ++t) {
    const int n = tn * 64 + t * 16 + lp;
    const float bias = bo[n];
#pragma unroll
    for (int r = 0; r < 8; ++r) {
      const int tok = tm * 16 + r + 8 * half;
      out[(size_t)tok * DMODEL + n] = acc[t][r] + bias;
    }
  }
}

// ------------------------------------------------------------- launcher -----
extern "C" void kernel_launch(void* const* d_in, const int* in_sizes, int n_in,
                              void* d_out, int out_size, void* d_ws, size_t ws_size,
                              hipStream_t stream) {
  const float* x  = (const float*)d_in[0];
  const unsigned char* amask = (const unsigned char*)d_in[1];
  const float* wq = (const float*)d_in[2];
  const float* wk = (const float*)d_in[3];
  const float* wv = (const float*)d_in[4];
  const float* wo = (const float*)d_in[5];
  const float* bo = (const float*)d_in[6];
  float* out = (float*)d_out;

  const size_t NX = (size_t)BATCH * SEQ * DMODEL;   // 4,194,304
  const size_t NW = (size_t)DMODEL * DMODEL;        // 1,048,576

  char* ws = (char*)d_ws;
  size_t off = 0;
  auto alloc = [&](size_t bytes) -> void* {
    void* p = ws + off;
    off = (off + bytes + 255) & ~(size_t)255;
    return p;
  };
  __bf16* x_bf  = (__bf16*)alloc(NX * 2);
  __bf16* wq_bf = (__bf16*)alloc(NW * 2);
  __bf16* wk_bf = (__bf16*)alloc(NW * 2);
  __bf16* wv_bf = (__bf16*)alloc(NW * 2);
  __bf16* wo_bf = (__bf16*)alloc(NW * 2);
  __bf16* q_bf  = (__bf16*)alloc(NX * 2);   // (b,h,s,hd)
  __bf16* k_bf  = (__bf16*)alloc(NX * 2);   // (b,h,s,hd)
  __bf16* vt_bf = (__bf16*)alloc(NX * 2);   // (b,h,hd,s)
  __bf16* ctx_bf= (__bf16*)alloc(NX * 2);   // (b,s,d)

  cvt_f32_bf16<<<2048, 256, 0, stream>>>(x,  x_bf,  (int)NX);
  cvt_f32_bf16<<<1024, 256, 0, stream>>>(wq, wq_bf, (int)NW);
  cvt_f32_bf16<<<1024, 256, 0, stream>>>(wk, wk_bf, (int)NW);
  cvt_f32_bf16<<<1024, 256, 0, stream>>>(wv, wv_bf, (int)NW);
  cvt_f32_bf16<<<1024, 256, 0, stream>>>(wo, wo_bf, (int)NW);

  dim3 gq(BATCH * SEQ / 16, DMODEL / HDIM, 3);
  qkv_kernel<<<gq, 32, 0, stream>>>(x_bf, wq_bf, wk_bf, wv_bf, q_bf, k_bf, vt_bf);

  dim3 ga(SEQ / 16, BATCH * NHEAD);
  attn_kernel<<<ga, 32, 0, stream>>>(q_bf, k_bf, vt_bf, amask, ctx_bf);

  dim3 go(BATCH * SEQ / 16, DMODEL / 64);
  oproj_kernel<<<go, 32, 0, stream>>>(ctx_bf, wo_bf, bo, out);
}
